// MultiLayerLSTMCell_60825326846606
// MI455X (gfx1250) — compile-verified, matrix operand order checked by reading, not run on hardware
//
#include <hip/hip_runtime.h>
#include <hip/hip_bf16.h>

typedef unsigned short u16;
typedef unsigned int   u32;
typedef __attribute__((ext_vector_type(16))) __bf16 v16bf;
typedef __attribute__((ext_vector_type(8)))  float  v8f;
typedef __attribute__((ext_vector_type(4)))  unsigned int v4u;

union Frag { v4u q[2]; v16bf v; };

#define HH   1024
#define BB   4096
#define LL   4
#define BM   128
#define BN   64
#define BK   32
#define LDST 40                    // LDS row stride in u16 (32 + 8 pad, 80B, 16B-aligned)
#define BUFSZ ((BM + 4 * BN) * LDST)   // 15360 u16 = 30 KB per buffer
#define NSTAGE 64                  // KTOT(2048) / BK

__device__ __forceinline__ u16 f32_to_bf16u(float f) {
    unsigned int u = __float_as_uint(f);
    unsigned int r = u + 0x7FFFu + ((u >> 16) & 1u);   // round-to-nearest-even
    return (u16)(r >> 16);
}

__device__ __forceinline__ float sigmoidf_(float x) {
    return 1.0f / (1.0f + __expf(-x));
}

// stable fast tanh: e = exp(-2|x|) in (0,1], no inf/inf path
__device__ __forceinline__ float tanh_fast(float x) {
    float e = __expf(-2.0f * fabsf(x));
    float r = (1.0f - e) / (1.0f + e);
    return copysignf(r, x);
}

// ---------------------------------------------------------------- conversions
__global__ void cvt_f32_bf16_kernel(const float* __restrict__ src,
                                    u16* __restrict__ dst, long n) {
    long stride = (long)gridDim.x * blockDim.x;
    for (long i = (long)blockIdx.x * blockDim.x + threadIdx.x; i < n; i += stride)
        dst[i] = f32_to_bf16u(src[i]);
}

// h0 is [B, H, L] f32; produce L contiguous bf16 planes of [B, H]
__global__ void deint_h0_kernel(const float* __restrict__ h0,
                                u16* __restrict__ dst, long total, long plane) {
    long stride = (long)gridDim.x * blockDim.x;
    for (long i = (long)blockIdx.x * blockDim.x + threadIdx.x; i < total; i += stride) {
        long l = i % LL;
        long j = i / LL;
        dst[l * plane + j] = f32_to_bf16u(h0[i]);
    }
}

// ---------------------------------------------------------------- fused layer
// gates[m, g*H+n] = sum_k in[m,k]*Wih[g*H+n,k] + h0[m,k]*Whh[g*H+n,k], then cell.
// Double-buffered LDS fed by GLOBAL_LOAD_ASYNC_TO_LDS_B128 (ASYNCcnt pipeline).
__global__ __launch_bounds__(256)
void lstm_layer_kernel(const u16* __restrict__ Wih,   // [4H, H] bf16 (this layer)
                       const u16* __restrict__ Whh,   // [4H, H] bf16
                       const float* __restrict__ bih, // [4H]
                       const float* __restrict__ bhh, // [4H]
                       const u16* __restrict__ hin,   // [B, H] bf16 (layer input)
                       const u16* __restrict__ h0b,   // [B, H] bf16 (this layer h0)
                       const float* __restrict__ c0,  // [B, H, L] f32 (full tensor)
                       float* __restrict__ outH,      // [B, H, L] f32
                       float* __restrict__ outC,      // [B, H, L] f32
                       u16* __restrict__ hnext,       // [B, H] bf16 (next layer in)
                       int layer) {
    __shared__ __align__(16) u16 lds[2 * BUFSZ];     // 60 KB double buffer

    const int tid  = threadIdx.x;
    const int wave = tid >> 5, lane = tid & 31;
    const int wm = wave >> 1, wn = wave & 1;       // 4 x 2 wave grid
    const int lh = lane >> 4, lr = lane & 15;      // half-wave, row-in-half
    const int gm0 = blockIdx.y * BM;               // batch offset
    const int gn0 = blockIdx.x * BN;               // hidden-unit offset

    v8f acc[4][2][2];
    #pragma unroll
    for (int g = 0; g < 4; ++g)
        #pragma unroll
        for (int mi = 0; mi < 2; ++mi)
            #pragma unroll
            for (int ni = 0; ni < 2; ++ni)
                acc[g][mi][ni] = v8f{0.f,0.f,0.f,0.f,0.f,0.f,0.f,0.f};

    // issue one BK=32 stage: 6 async b128 copies per wave (A: 2, B: 4 per thread)
    auto issue = [&](int s, int bi) {
        const u16* aSrc = (s < 32) ? hin : h0b;
        const u16* wSrc = (s < 32) ? Wih : Whh;
        const int  kk   = (s & 31) * BK;
        u16* bufA = lds + bi * BUFSZ;
        u16* bufB = bufA + BM * LDST;
        #pragma unroll
        for (int t = 0; t < 2; ++t) {                 // A tile: 128 x 32 bf16
            int u = tid + t * 256;
            int row = u >> 2, ck = (u & 3) * 8;
            const u16* g = aSrc + (size_t)(gm0 + row) * HH + kk + ck;
            u32 ldsoff = (u32)(size_t)(bufA + row * LDST + ck);
            asm volatile("global_load_async_to_lds_b128 %0, %1, off"
                         :: "v"(ldsoff), "v"((unsigned long long)(size_t)g) : "memory");
        }
        #pragma unroll
        for (int t = 0; t < 4; ++t) {                 // B tiles: 4 x 64 x 32 bf16
            int u = tid + t * 256;
            int row = u >> 2, ck = (u & 3) * 8;       // row = gate*64 + n
            int gate = row >> 6, n = row & 63;
            const u16* g = wSrc + (size_t)(gate * HH + gn0 + n) * HH + kk + ck;
            u32 ldsoff = (u32)(size_t)(bufB + row * LDST + ck);
            asm volatile("global_load_async_to_lds_b128 %0, %1, off"
                         :: "v"(ldsoff), "v"((unsigned long long)(size_t)g) : "memory");
        }
    };

    issue(0, 0);

    for (int s = 0; s < NSTAGE; ++s) {
        if (s < NSTAGE - 1) {
            issue(s + 1, (s + 1) & 1);                // prefetch next stage
            asm volatile("s_wait_asynccnt 6" ::: "memory");   // stage s arrived (mine)
        } else {
            asm volatile("s_wait_asynccnt 0" ::: "memory");
        }
        __syncthreads();                              // everyone's stage s arrived

        const u16* bufA = lds + (s & 1) * BUFSZ;
        const u16* bufB = bufA + BM * LDST;
        const int  kb   = lh * 8;                     // per-lane K chunk base

        Frag a0, a1;
        {
            const u16* p0 = bufA + (wm * 32 + lr) * LDST + kb;
            a0.q[0] = *(const v4u*)(p0);
            a0.q[1] = *(const v4u*)(p0 + 16);
            const u16* p1 = bufA + (wm * 32 + 16 + lr) * LDST + kb;
            a1.q[0] = *(const v4u*)(p1);
            a1.q[1] = *(const v4u*)(p1 + 16);
        }
        #pragma unroll
        for (int g = 0; g < 4; ++g) {
            Frag b0, b1;
            const u16* p0 = bufB + (g * 64 + wn * 32 + lr) * LDST + kb;
            b0.q[0] = *(const v4u*)(p0);
            b0.q[1] = *(const v4u*)(p0 + 16);
            const u16* p1 = bufB + (g * 64 + wn * 32 + 16 + lr) * LDST + kb;
            b1.q[0] = *(const v4u*)(p1);
            b1.q[1] = *(const v4u*)(p1 + 16);

            acc[g][0][0] = __builtin_amdgcn_wmma_f32_16x16x32_bf16(
                false, a0.v, false, b0.v, (short)0, acc[g][0][0], false, false);
            acc[g][0][1] = __builtin_amdgcn_wmma_f32_16x16x32_bf16(
                false, a0.v, false, b1.v, (short)0, acc[g][0][1], false, false);
            acc[g][1][0] = __builtin_amdgcn_wmma_f32_16x16x32_bf16(
                false, a1.v, false, b0.v, (short)0, acc[g][1][0], false, false);
            acc[g][1][1] = __builtin_amdgcn_wmma_f32_16x16x32_bf16(
                false, a1.v, false, b1.v, (short)0, acc[g][1][1], false, false);
        }
        __syncthreads();   // all waves done reading buf[s&1] before stage s+2 refills it
    }

    // ---------------- epilogue: bias + LSTM cell ----------------
    float bs[4][2];
    #pragma unroll
    for (int g = 0; g < 4; ++g)
        #pragma unroll
        for (int ni = 0; ni < 2; ++ni) {
            int n = gn0 + wn * 32 + ni * 16 + lr;
            bs[g][ni] = bih[g * HH + n] + bhh[g * HH + n];
        }

    const float* c0p = c0 + layer;
    float* oh = outH + layer;
    float* oc = outC + layer;

    #pragma unroll
    for (int mi = 0; mi < 2; ++mi)
        #pragma unroll
        for (int ni = 0; ni < 2; ++ni) {
            int n = gn0 + wn * 32 + ni * 16 + lr;
            #pragma unroll
            for (int r = 0; r < 8; ++r) {
                int m = gm0 + wm * 32 + mi * 16 + lh * 8 + r;
                float gi = sigmoidf_(acc[0][mi][ni][r] + bs[0][ni]);
                float gf = sigmoidf_(acc[1][mi][ni][r] + bs[1][ni]);
                float gg = tanh_fast(acc[2][mi][ni][r] + bs[2][ni]);
                float go = sigmoidf_(acc[3][mi][ni][r] + bs[3][ni]);
                size_t idx = (size_t)m * HH + n;
                float cc = gf * c0p[idx * LL] + gi * gg;
                float hh = go * tanh_fast(cc);
                oh[idx * LL] = hh;
                oc[idx * LL] = cc;
                hnext[idx]   = f32_to_bf16u(hh);
            }
        }
}

// ---------------------------------------------------------------- launch
extern "C" void kernel_launch(void* const* d_in, const int* in_sizes, int n_in,
                              void* d_out, int out_size, void* d_ws, size_t ws_size,
                              hipStream_t stream) {
    const float* x    = (const float*)d_in[0];
    const float* h0   = (const float*)d_in[1];
    const float* c0   = (const float*)d_in[2];
    const float* W_ih = (const float*)d_in[3];
    const float* W_hh = (const float*)d_in[4];
    const float* b_ih = (const float*)d_in[5];
    const float* b_hh = (const float*)d_in[6];

    const long BH   = (long)BB * HH;          // 4,194,304
    const long BHL  = BH * LL;                // 16,777,216
    const long WEL  = (long)LL * 4 * HH * HH; // 16,777,216 (all layers)
    const long WPL  = (long)4 * HH * HH;      // per-layer weight elems

    // workspace layout (bf16 = u16): Wih | Whh | h0 planes | ping | pong
    u16* wih_b = (u16*)d_ws;
    u16* whh_b = wih_b + WEL;
    u16* h0_b  = whh_b + WEL;
    u16* hb0   = h0_b  + BHL;
    u16* hb1   = hb0   + BH;

    float* outH = (float*)d_out;
    float* outC = outH + BHL;

    const int CVB = 8192;
    cvt_f32_bf16_kernel<<<CVB, 256, 0, stream>>>(W_ih, wih_b, WEL);
    cvt_f32_bf16_kernel<<<CVB, 256, 0, stream>>>(W_hh, whh_b, WEL);
    cvt_f32_bf16_kernel<<<CVB, 256, 0, stream>>>(x,    hb0,   BH);
    deint_h0_kernel    <<<CVB, 256, 0, stream>>>(h0,   h0_b,  BHL, BH);

    dim3 grid(HH / BN, BB / BM);   // 16 x 32
    for (int l = 0; l < LL; ++l) {
        u16* hin  = (l & 1) ? hb1 : hb0;
        u16* hout = (l & 1) ? hb0 : hb1;
        lstm_layer_kernel<<<grid, 256, 0, stream>>>(
            wih_b + (size_t)l * WPL, whh_b + (size_t)l * WPL,
            b_ih + (size_t)l * 4 * HH, b_hh + (size_t)l * 4 * HH,
            hin, h0_b + (size_t)l * BH, c0,
            outH, outC, hout, l);
    }
}
